// ContextConditionedAttention_64355789963685
// MI455X (gfx1250) — matrix operand order repaired and checked by reading, not compiled, not run on hardware
//
#include <hip/hip_runtime.h>
#include <hip/hip_bf16.h>

// ---------------------------------------------------------------------------
// ContextConditionedAttention for MI455X (gfx1250, wave32, WMMA + TDM)
//   B=4, N=2048, HIDDEN=512, H=8, D=64, scale = 1/sqrt(64) = 0.125
// Pipeline:
//   k1: Q/K/V projection, 32x64 tile/wave, 8 WMMA per K-step, +bias/+keymod
//   k2: flash attention, V tiles staged to LDS by the Tensor Data Mover
//       (double-buffered, s_wait_tensorcnt), f32 online softmax
//   k3: output projection, 32x64 tile/wave, f32 result to d_out
// ---------------------------------------------------------------------------

typedef _Float16 half_t;
typedef __attribute__((ext_vector_type(16))) _Float16 v16h;
typedef __attribute__((ext_vector_type(8)))  float    v8f;

#define WMMA_F16(A, B, C)                                                  \
  __builtin_amdgcn_wmma_f32_16x16x32_f16(false, (A), false, (B), (short)0, \
                                         (C), false, false)

#define NB 4
#define NSEQ 2048
#define HID 512
#define NH 8
#define DH 64
#define VLD 66  // LDS row stride (halves) for V tiles: 32 dw + 1 dw TDM pad

#if defined(__has_builtin)
#if __has_builtin(__builtin_amdgcn_tensor_load_to_lds) && \
    __has_builtin(__builtin_amdgcn_s_wait_tensorcnt)
#define USE_TDM 1
#endif
#endif
#ifndef USE_TDM
#define USE_TDM 0
#endif

__device__ __forceinline__ v8f zero_v8f() {
  v8f c;
#pragma unroll
  for (int i = 0; i < 8; ++i) c[i] = 0.0f;
  return c;
}

// A-fragment (16x32 f16): lane m = lane&15 is row; half hh = lane>>4.
// a[i] = src[m][hh*16 + i], i = 0..15
__device__ __forceinline__ v16h afrag_f32(const float* src, int ld) {
  const int lane = threadIdx.x & 31;
  const int m = lane & 15, hh = lane >> 4;
  const float* p = src + (size_t)m * ld + hh * 16;
  v16h a;
#pragma unroll
  for (int i = 0; i < 8; ++i) a[i] = (_Float16)p[i];
#pragma unroll
  for (int i = 0; i < 8; ++i) a[8 + i] = (_Float16)p[8 + i];
  return a;
}

__device__ __forceinline__ v16h afrag_f16(const half_t* src, int ld) {
  const int lane = threadIdx.x & 31;
  const int m = lane & 15, hh = lane >> 4;
  const half_t* p = src + (size_t)m * ld + hh * 16;
  v16h a;
#pragma unroll
  for (int i = 0; i < 8; ++i) a[i] = p[i];
#pragma unroll
  for (int i = 0; i < 8; ++i) a[8 + i] = p[8 + i];
  return a;
}

// B-fragment (32x16) with B[k][n] = src[n*ld + k] (transposed row-major src).
__device__ __forceinline__ v16h bfragT_f32(const float* src, int ld) {
  const int lane = threadIdx.x & 31;
  const int n = lane & 15, hh = lane >> 4;
  const float* p = src + (size_t)n * ld + hh * 16;
  v16h b;
#pragma unroll
  for (int i = 0; i < 16; ++i) b[i] = (_Float16)p[i];
  return b;
}

__device__ __forceinline__ v16h bfragT_f16(const half_t* src, int ld) {
  const int lane = threadIdx.x & 31;
  const int n = lane & 15, hh = lane >> 4;
  const half_t* p = src + (size_t)n * ld + hh * 16;
  v16h b;
#pragma unroll
  for (int i = 0; i < 16; ++i) b[i] = p[i];
  return b;
}

// B-fragment (32x16) with B[k][n] = src[k*ld + n] (row-major, row = K).
__device__ __forceinline__ v16h bfragR_f16(const _Float16* src, int ld) {
  const int lane = threadIdx.x & 31;
  const int n = lane & 15, hh = lane >> 4;
  const _Float16* p = src + (size_t)(hh * 16) * ld + n;
  v16h b;
#pragma unroll
  for (int i = 0; i < 16; ++i) b[i] = p[(size_t)i * ld];
  return b;
}

__device__ __forceinline__ float redmax16(float v) {
  v = fmaxf(v, __shfl_xor(v, 1, 32));
  v = fmaxf(v, __shfl_xor(v, 2, 32));
  v = fmaxf(v, __shfl_xor(v, 4, 32));
  v = fmaxf(v, __shfl_xor(v, 8, 32));
  return v;
}
__device__ __forceinline__ float redsum16(float v) {
  v += __shfl_xor(v, 1, 32);
  v += __shfl_xor(v, 2, 32);
  v += __shfl_xor(v, 4, 32);
  v += __shfl_xor(v, 8, 32);
  return v;
}

#if USE_TDM
typedef __attribute__((ext_vector_type(4))) unsigned int v4u_t;
typedef __attribute__((ext_vector_type(8))) int v8i_t;
typedef __attribute__((ext_vector_type(4))) int v4i_t;

// TDM: DMA a 32x64 f16 tile (row stride 64) from global into LDS, inserting
// 1 dword of padding every 32 dwords (=> LDS row stride 66 halves).
// D# encoding per CDNA5 ISA ch.8 (group0: count/lds/global/type,
// group1: data_size, pad, dims, strides; groups 2/3 unused for 2D tiles).
// This toolchain exposes the 6-arg builtin form:
//   (uint32x4 g0, int32x8 g1, int32x4 g2, int32x4 g3, int32x8 g4, i32 cpol)
__device__ __forceinline__ void tdm_load_v_tile(const half_t* gsrc,
                                                unsigned lds_byte) {
  const unsigned long long ga = (unsigned long long)(uintptr_t)gsrc;
  v4u_t g0;
  g0[0] = 1u;                            // count=1 (valid user descriptor)
  g0[1] = lds_byte;                      // lds_addr (bytes)
  g0[2] = (unsigned)ga;                  // global_addr[31:0]
  g0[3] = (unsigned)((ga >> 32) & 0x01FFFFFFu) | 0x80000000u;  // [56:32]|type=2
  v8i_t g1;
  g1[0] = (1 << 16) | (1 << 20) | (4 << 22);  // data_size=2B|pad_en|pad@32dw(+1dw)
  g1[1] = (int)((unsigned)DH << 16);     // tensor_dim0 = 64 (bits 63:48)
  g1[2] = (int)((unsigned)NSEQ << 16);   // tensor_dim1 = 2048 (bits 95:80)
  g1[3] = (int)((unsigned)DH << 16);     // tile_dim0 = 64 (bits 127:112)
  g1[4] = 32;                            // tile_dim1 = 32 rows, tile_dim2 = 0
  g1[5] = DH;                            // tensor_dim0_stride = 64
  g1[6] = 0;
  g1[7] = 0;
  v4i_t gz4;
#pragma unroll
  for (int i = 0; i < 4; ++i) gz4[i] = 0;
  v8i_t gz8;
#pragma unroll
  for (int i = 0; i < 8; ++i) gz8[i] = 0;
  __builtin_amdgcn_tensor_load_to_lds(g0, g1, gz4, gz4, gz8, 0);
}
#else
// Fallback: lane-parallel synchronous copy global -> LDS (row stride VLD).
__device__ __forceinline__ void stage_v_sync(const half_t* gsrc,
                                             _Float16* dst) {
  const int lane = threadIdx.x & 31;
  const half_t* s = gsrc + (size_t)lane * DH;
  _Float16* d = dst + lane * VLD;
#pragma unroll
  for (int i = 0; i < DH; ++i) d[i] = s[i];
  __syncthreads();
}
#endif

// ---------------------------------------------------------------------------
// Kernel 1: QKV projection. grid = (HID/64, (B*N)/32, 3), block = 32.
// Each wave: 32 rows x 64 cols (one full head slice) = 8 accumulators.
// out[r][o] = sum_k x[r][k]*W[o][k] + bias[o] (+ keymod[ct[b]][o] for K)
// ---------------------------------------------------------------------------
__global__ void __launch_bounds__(32)
qkv_proj_kernel(const float* __restrict__ x,
                const float* __restrict__ wq, const float* __restrict__ bq,
                const float* __restrict__ wk, const float* __restrict__ bk,
                const float* __restrict__ wv, const float* __restrict__ bv,
                const float* __restrict__ keymod, const int* __restrict__ ct,
                half_t* __restrict__ Qh, half_t* __restrict__ Kh,
                half_t* __restrict__ Vh) {
  const int o0 = blockIdx.x * 64;   // 64-aligned -> single head
  const int r0 = blockIdx.y * 32;
  const int mat = blockIdx.z;
  const int b = r0 >> 11;  // 2048 rows per batch

  const float* W = (mat == 0) ? wq : (mat == 1) ? wk : wv;
  const float* bias = (mat == 0) ? bq : (mat == 1) ? bk : bv;
  half_t* out = (mat == 0) ? Qh : (mat == 1) ? Kh : Vh;

  v8f c[2][4];
#pragma unroll
  for (int rr = 0; rr < 2; ++rr)
#pragma unroll
    for (int t = 0; t < 4; ++t) c[rr][t] = zero_v8f();

#pragma unroll 2
  for (int ks = 0; ks < HID; ks += 32) {
    const v16h a0 = afrag_f32(x + (size_t)r0 * HID + ks, HID);
    const v16h a1 = afrag_f32(x + (size_t)(r0 + 16) * HID + ks, HID);
#pragma unroll
    for (int t = 0; t < 4; ++t) {
      const v16h bb = bfragT_f32(W + (size_t)(o0 + 16 * t) * HID + ks, HID);
      c[0][t] = WMMA_F16(a0, bb, c[0][t]);
      c[1][t] = WMMA_F16(a1, bb, c[1][t]);
    }
  }

  const int lane = threadIdx.x & 31;
  const int n = lane & 15, hh = lane >> 4;
  const int head = o0 >> 6;
  half_t* obase = out + ((size_t)b * NH + head) * NSEQ * DH;
#pragma unroll
  for (int t = 0; t < 4; ++t) {
    const int o = o0 + t * 16 + n;
    float add = bias[o];
    if (mat == 1) add += keymod[(size_t)ct[b] * HID + o];
    const int d = t * 16 + n;
#pragma unroll
    for (int rr = 0; rr < 2; ++rr)
#pragma unroll
      for (int v = 0; v < 8; ++v) {
        const int r = r0 + rr * 16 + v + 8 * hh;
        const int nn = r & (NSEQ - 1);
        obase[(size_t)nn * DH + d] = (_Float16)(c[rr][t][v] + add);
      }
  }
}

// ---------------------------------------------------------------------------
// Kernel 2: flash attention. grid = (N/16, H, B), block = 32 (1 wave).
// S = Q K^T / 8 + bias(b,h); mask -> -inf; online softmax; O = P V.
// V tiles streamed into LDS by TDM (double-buffered) when available.
// ---------------------------------------------------------------------------
__global__ void __launch_bounds__(32)
attn_kernel(const half_t* __restrict__ Qh, const half_t* __restrict__ Kh,
            const half_t* __restrict__ Vh,
            const unsigned char* __restrict__ mask,
            const float* __restrict__ bias_emb, const int* __restrict__ ct,
            half_t* __restrict__ attnh) {
  __shared__ _Float16 P[16][34];        // 16x32 probability tile (+pad)
  __shared__ _Float16 Vt[2][32 * VLD];  // double-buffered 32x64 V tiles

  const int i0 = blockIdx.x * 16;
  const int head = blockIdx.y;
  const int b = blockIdx.z;
  const int lane = threadIdx.x & 31;
  const int n = lane & 15, hh = lane >> 4;

  const half_t* Qbh = Qh + ((size_t)b * NH + head) * NSEQ * DH;
  const half_t* Kbh = Kh + ((size_t)b * NH + head) * NSEQ * DH;
  const half_t* Vbh = Vh + ((size_t)b * NH + head) * NSEQ * DH;
  const unsigned char* mb = mask + (size_t)b * NSEQ;
  const float cbias = bias_emb[(size_t)ct[b] * NH + head];

  const v16h q0 = afrag_f16(Qbh + (size_t)i0 * DH + 0, DH);
  const v16h q1 = afrag_f16(Qbh + (size_t)i0 * DH + 32, DH);

  float mrow[8], lrow[8];
  v8f oacc[4];
#pragma unroll
  for (int v = 0; v < 8; ++v) { mrow[v] = -3.0e38f; lrow[v] = 0.0f; }
#pragma unroll
  for (int t = 0; t < 4; ++t) oacc[t] = zero_v8f();

#if USE_TDM
  tdm_load_v_tile(Vbh, (unsigned)(uintptr_t)&Vt[0][0]);
#endif

  for (int j = 0; j < NSEQ; j += 32) {
    const int cur = (j >> 5) & 1;
#if USE_TDM
    if (j + 32 < NSEQ)  // prefetch next V tile while we do the QK^T WMMAs
      tdm_load_v_tile(Vbh + (size_t)(j + 32) * DH,
                      (unsigned)(uintptr_t)&Vt[cur ^ 1][0]);
#else
    stage_v_sync(Vbh + (size_t)j * DH, &Vt[cur][0]);
#endif

    // ---- scores for 32 key columns (two 16x16 C tiles) ----
    v8f s0 = zero_v8f(), s1 = zero_v8f();
    {
      v16h kb = bfragT_f16(Kbh + (size_t)j * DH + 0, DH);
      s0 = WMMA_F16(q0, kb, s0);
      kb = bfragT_f16(Kbh + (size_t)j * DH + 32, DH);
      s0 = WMMA_F16(q1, kb, s0);
      kb = bfragT_f16(Kbh + (size_t)(j + 16) * DH + 0, DH);
      s1 = WMMA_F16(q0, kb, s1);
      kb = bfragT_f16(Kbh + (size_t)(j + 16) * DH + 32, DH);
      s1 = WMMA_F16(q1, kb, s1);
    }
    const bool msk0 = mb[j + n] != 0;
    const bool msk1 = mb[j + 16 + n] != 0;

    // ---- online softmax update (rows in VGPR index, cols in lanes) ----
#pragma unroll
    for (int v = 0; v < 8; ++v) {
      float a0 = s0[v] * 0.125f + cbias;
      float a1 = s1[v] * 0.125f + cbias;
      if (msk0) a0 = -3.0e38f;
      if (msk1) a1 = -3.0e38f;
      const float rm = redmax16(fmaxf(a0, a1));
      const float mnew = fmaxf(mrow[v], rm);
      const float alpha = __expf(mrow[v] - mnew);
      mrow[v] = mnew;
      const float p0 = __expf(a0 - mnew);
      const float p1 = __expf(a1 - mnew);
      lrow[v] = lrow[v] * alpha + redsum16(p0 + p1);
#pragma unroll
      for (int t = 0; t < 4; ++t) oacc[t][v] *= alpha;
      P[v + 8 * hh][n] = (_Float16)p0;
      P[v + 8 * hh][16 + n] = (_Float16)p1;
    }
    __syncthreads();
    const v16h pf = afrag_f16(&P[0][0], 34);
    __syncthreads();

#if USE_TDM
    if (j + 32 < NSEQ) __builtin_amdgcn_s_wait_tensorcnt(1);
    else               __builtin_amdgcn_s_wait_tensorcnt(0);
#endif

    // ---- O += P(16x32) x V(32x64): four 16x16 output tiles ----
    {
      const _Float16* vt = &Vt[cur][0];
      v16h vb = bfragR_f16(vt + 0, VLD);
      oacc[0] = WMMA_F16(pf, vb, oacc[0]);
      vb = bfragR_f16(vt + 16, VLD);
      oacc[1] = WMMA_F16(pf, vb, oacc[1]);
      vb = bfragR_f16(vt + 32, VLD);
      oacc[2] = WMMA_F16(pf, vb, oacc[2]);
      vb = bfragR_f16(vt + 48, VLD);
      oacc[3] = WMMA_F16(pf, vb, oacc[3]);
    }
  }

  // ---- normalize, store to (B, N, HIDDEN) f16 for the output projection
#pragma unroll
  for (int v = 0; v < 8; ++v) {
    const float inv = 1.0f / lrow[v];
    const int r = i0 + v + 8 * hh;
    const size_t base = ((size_t)b * NSEQ + r) * HID + head * DH;
#pragma unroll
    for (int t = 0; t < 4; ++t)
      attnh[base + t * 16 + n] = (_Float16)(oacc[t][v] * inv);
  }
}

// ---------------------------------------------------------------------------
// Kernel 3: output projection. grid = (HID/64, (B*N)/32), block = 32.
// d_out[r][o] = sum_k attnh[r][k] * wo[o][k] + bo[o]   (f32 result)
// ---------------------------------------------------------------------------
__global__ void __launch_bounds__(32)
out_proj_kernel(const half_t* __restrict__ attnh, const float* __restrict__ wo,
                const float* __restrict__ bo, float* __restrict__ out) {
  const int o0 = blockIdx.x * 64;
  const int r0 = blockIdx.y * 32;

  v8f c[2][4];
#pragma unroll
  for (int rr = 0; rr < 2; ++rr)
#pragma unroll
    for (int t = 0; t < 4; ++t) c[rr][t] = zero_v8f();

#pragma unroll 2
  for (int ks = 0; ks < HID; ks += 32) {
    const v16h a0 = afrag_f16(attnh + (size_t)r0 * HID + ks, HID);
    const v16h a1 = afrag_f16(attnh + (size_t)(r0 + 16) * HID + ks, HID);
#pragma unroll
    for (int t = 0; t < 4; ++t) {
      const v16h bb = bfragT_f32(wo + (size_t)(o0 + 16 * t) * HID + ks, HID);
      c[0][t] = WMMA_F16(a0, bb, c[0][t]);
      c[1][t] = WMMA_F16(a1, bb, c[1][t]);
    }
  }

  const int lane = threadIdx.x & 31;
  const int n = lane & 15, hh = lane >> 4;
#pragma unroll
  for (int t = 0; t < 4; ++t) {
    const int o = o0 + t * 16 + n;
    const float add = bo[o];
#pragma unroll
    for (int rr = 0; rr < 2; ++rr)
#pragma unroll
      for (int v = 0; v < 8; ++v) {
        const int r = r0 + rr * 16 + v + 8 * hh;
        out[(size_t)r * HID + o] = c[rr][t][v] + add;
      }
  }
}

// ---------------------------------------------------------------------------
extern "C" void kernel_launch(void* const* d_in, const int* in_sizes, int n_in,
                              void* d_out, int out_size, void* d_ws,
                              size_t ws_size, hipStream_t stream) {
  const float* x = (const float*)d_in[0];
  const int* ct = (const int*)d_in[1];
  const unsigned char* mask = (const unsigned char*)d_in[2];
  const float* wq = (const float*)d_in[3];
  const float* bq = (const float*)d_in[4];
  const float* wk = (const float*)d_in[5];
  const float* bk = (const float*)d_in[6];
  const float* wv = (const float*)d_in[7];
  const float* bv = (const float*)d_in[8];
  const float* wo = (const float*)d_in[9];
  const float* bo = (const float*)d_in[10];
  const float* bias_emb = (const float*)d_in[11];
  const float* keymod = (const float*)d_in[12];
  float* out = (float*)d_out;

  // workspace: Q, K, V (B,H,N,DH f16) + attn output (B,N,HID f16) = 32 MiB
  const size_t qkv_elems = (size_t)NB * NH * NSEQ * DH;  // 4,194,304
  half_t* Qh = (half_t*)d_ws;
  half_t* Kh = Qh + qkv_elems;
  half_t* Vh = Kh + qkv_elems;
  half_t* Ah = Vh + qkv_elems;

  dim3 blk(32, 1, 1);

  dim3 g1(HID / 64, (NB * NSEQ) / 32, 3);  // 8 x 256 x 3
  qkv_proj_kernel<<<g1, blk, 0, stream>>>(x, wq, bq, wk, bk, wv, bv, keymod, ct,
                                          Qh, Kh, Vh);

  dim3 g2(NSEQ / 16, NH, NB);  // 128 x 8 x 4
  attn_kernel<<<g2, blk, 0, stream>>>(Qh, Kh, Vh, mask, bias_emb, ct, Ah);

  dim3 g3(HID / 64, (NB * NSEQ) / 32, 1);  // 8 x 256
  out_proj_kernel<<<g3, blk, 0, stream>>>(Ah, wo, bo, out);
}
